// SingleHeadAttention_8675833938647
// MI455X (gfx1250) — matrix operand order repaired and checked by reading, not compiled, not run on hardware
//
#include <hip/hip_runtime.h>
#include <hip/hip_bf16.h>

// ---------------------------------------------------------------------------
// Single-head causal attention for MI455X (gfx1250), bf16 WMMA path.
//   B=8, S=2048, E=1024, D=64.  All matmuls via v_wmma_f32_16x16x32_bf16.
//   Phase 2 stages K/V chunks into LDS with GLOBAL_LOAD_ASYNC_TO_LDS_B128,
//   double-buffered, synchronized with s_wait_asynccnt + barriers.
// Workspace: q bf16 [B*S,64] | k bf16 [B*S,64] | vT bf16 [B,64,S]  (6 MB).
// ---------------------------------------------------------------------------

typedef __attribute__((ext_vector_type(16))) __bf16 v16bf;
typedef __attribute__((ext_vector_type(8)))  float  v8f;

#define BATCH 8
#define SEQ   2048
#define EMB   1024
#define HD    64
#define MROWS (BATCH * SEQ)

union Frag16 { v16bf v; uint4 q[2]; };

// A-matrix 16x32 bf16 fragment (ISA 7.12.2):
//   lane<16 : M=lane,    K = {0..7, 16..23}
//   lane>=16: M=lane-16, K = {8..15, 24..31}
__device__ __forceinline__ v16bf a_frag_bf16(const __bf16* base, int ldk, int lane) {
  int m    = lane & 15;
  int koff = (lane & 16) ? 8 : 0;
  Frag16 f;
  f.q[0] = *(const uint4*)(base + m * ldk + koff);
  f.q[1] = *(const uint4*)(base + m * ldk + koff + 16);
  return f.v;
}

// B-matrix 32x16 bf16 fragment: element (k,n) at base[n*ldn + k]
//   lane<16 : N=lane,    K = {0..7, 16..23}
//   lane>=16: N=lane-16, K = {8..15, 24..31}
__device__ __forceinline__ v16bf b_frag_bf16(const __bf16* base, int ldn, int lane) {
  int n    = lane & 15;
  int koff = (lane & 16) ? 8 : 0;
  Frag16 f;
  f.q[0] = *(const uint4*)(base + n * ldn + koff);
  f.q[1] = *(const uint4*)(base + n * ldn + koff + 16);
  return f.v;
}

// A fragment built from fp32 source (converted to bf16 on the fly).
__device__ __forceinline__ v16bf a_frag_f32(const float* base, int ldk, int lane) {
  int m    = lane & 15;
  int koff = (lane & 16) ? 8 : 0;
  const float* p = base + m * ldk + koff;
  float4 f0 = *(const float4*)(p + 0);
  float4 f1 = *(const float4*)(p + 4);
  float4 f2 = *(const float4*)(p + 16);
  float4 f3 = *(const float4*)(p + 20);
  v16bf r;
  r[0]  = (__bf16)f0.x; r[1]  = (__bf16)f0.y; r[2]  = (__bf16)f0.z; r[3]  = (__bf16)f0.w;
  r[4]  = (__bf16)f1.x; r[5]  = (__bf16)f1.y; r[6]  = (__bf16)f1.z; r[7]  = (__bf16)f1.w;
  r[8]  = (__bf16)f2.x; r[9]  = (__bf16)f2.y; r[10] = (__bf16)f2.z; r[11] = (__bf16)f2.w;
  r[12] = (__bf16)f3.x; r[13] = (__bf16)f3.y; r[14] = (__bf16)f3.z; r[15] = (__bf16)f3.w;
  return r;
}

__device__ __forceinline__ v8f wmma_bf16(v16bf a, v16bf b, v8f c) {
  return __builtin_amdgcn_wmma_f32_16x16x32_bf16(false, a, false, b, (short)0, c,
                                                 false, false);
}

__device__ __forceinline__ float fast_exp2(float x) {
#if __has_builtin(__builtin_amdgcn_exp2f)
  return __builtin_amdgcn_exp2f(x);
#else
  return exp2f(x);
#endif
}

// Async global->LDS copy, 16 bytes per lane (tracked by ASYNCcnt).
// LDS operand = low 32 bits of the generic pointer (== wave-relative offset).
__device__ __forceinline__ void async_g2l_b128(const void* gptr, void* lptr) {
  unsigned l = (unsigned)(unsigned long long)lptr;
  asm volatile("global_load_async_to_lds_b128 %0, %1, off"
               :: "v"(l), "v"(gptr)
               : "memory");
}

// ---------------------------------------------------------------------------
// Phase 1: QKV projection.  grid = (128, 1, 3); block = 256 (8 waves).
//   z=0 -> Q, z=1 -> K (both bf16 [B*S,64]); z=2 -> V transposed bf16 [B,64,S].
// W [E,64] fp32 is staged through LDS as bf16, transposed to [64][Kchunk].
// ---------------------------------------------------------------------------
#define WKC 256      // K-chunk of W kept in LDS
#define WTP 264      // padded row stride (elements) -> conflict-free b128 reads

__global__ __launch_bounds__(256)
void qkv_proj_kernel(const float* __restrict__ x,
                     const float* __restrict__ Wq,
                     const float* __restrict__ Wk,
                     const float* __restrict__ Wv,
                     __bf16* __restrict__ qo,
                     __bf16* __restrict__ ko,
                     __bf16* __restrict__ vto) {
  __shared__ __bf16 wt[64 * WTP];           // 33 KB
  const int tid  = threadIdx.x;
  const int lane = tid & 31;
  const int wave = tid >> 5;
  const int which = blockIdx.z;
  const float* W = (which == 0) ? Wq : (which == 1) ? Wk : Wv;

  const int row0 = (blockIdx.x * 8 + wave) * 16;     // 16-row M tile per wave
  const float* xrow = x + (size_t)row0 * EMB;

  v8f acc[4];
  #pragma unroll
  for (int i = 0; i < 4; ++i)
    #pragma unroll
    for (int r = 0; r < 8; ++r) acc[i][r] = 0.0f;

  for (int kc = 0; kc < EMB; kc += WKC) {
    __syncthreads();                                 // protect previous chunk
    // cooperative fill: W[k][n] fp32 -> wt[n][k] bf16 (transposed)
    for (int i = tid; i < 64 * WKC; i += 256) {
      int kl = i >> 6, n = i & 63;
      wt[n * WTP + kl] = (__bf16)W[(size_t)(kc + kl) * HD + n];
    }
    __syncthreads();

    #pragma unroll 2
    for (int k0 = 0; k0 < WKC; k0 += 32) {
      v16bf a = a_frag_f32(xrow + kc + k0, EMB, lane);
      #pragma unroll
      for (int nt = 0; nt < 4; ++nt) {
        v16bf b = b_frag_bf16(&wt[nt * 16 * WTP + k0], WTP, lane);
        acc[nt] = wmma_bf16(a, b, acc[nt]);
      }
    }
  }

  // epilogue: C layout is row = r + 8*(lane>=16), col = lane%16 (per N tile)
  const int myn = lane & 15;
  const int hi  = (lane & 16) ? 8 : 0;
  if (which == 2) {
    const int b  = row0 >> 11;          // /SEQ
    const int s0 = row0 & (SEQ - 1);
    __bf16* vb = vto + (size_t)b * HD * SEQ;
    #pragma unroll
    for (int nt = 0; nt < 4; ++nt)
      #pragma unroll
      for (int r = 0; r < 8; ++r)
        vb[(size_t)(nt * 16 + myn) * SEQ + s0 + r + hi] = (__bf16)acc[nt][r];
  } else {
    __bf16* o = ((which == 0) ? qo : ko) + (size_t)row0 * HD;
    #pragma unroll
    for (int nt = 0; nt < 4; ++nt)
      #pragma unroll
      for (int r = 0; r < 8; ++r)
        o[(size_t)(r + hi) * HD + nt * 16 + myn] = (__bf16)acc[nt][r];
  }
}

// ---------------------------------------------------------------------------
// Phase 2: causal flash attention.
// Block = 8 waves = 8 consecutive 16-query tiles (same batch).
// K/V 32-key chunks staged once per block into LDS via async b128 copies,
// double-buffered; each wave consumes them as WMMA B-fragments.
// ---------------------------------------------------------------------------
#define KTP 72   // padded LDS stride for K chunk rows (keys) [32][64]
#define VTP 40   // padded LDS stride for V chunk rows (dims) [64][32]

// Stage one 32-key chunk: K rows [j..j+31] x 64 dims, vT 64 dims x keys[j..j+31].
// 256 threads x 2 async b128 each = 8 KB.
__device__ __forceinline__ void issue_chunk(const __bf16* kb, const __bf16* vtb,
                                            int j, __bf16* kbuf, __bf16* vbuf,
                                            int tid) {
  {
    int row = tid >> 3, seg = tid & 7;     // 32 rows x 8 segs of 16B
    async_g2l_b128(kb + (size_t)(j + row) * HD + seg * 8,
                   kbuf + row * KTP + seg * 8);
  }
  {
    int dim = tid >> 2, seg = tid & 3;     // 64 dims x 4 segs of 16B
    async_g2l_b128(vtb + (size_t)dim * SEQ + j + seg * 8,
                   vbuf + dim * VTP + seg * 8);
  }
}

__global__ __launch_bounds__(256)
void attn_kernel(const __bf16* __restrict__ q,
                 const __bf16* __restrict__ k,
                 const __bf16* __restrict__ vt,
                 float* __restrict__ out) {
  __shared__ __bf16 kls[2][32 * KTP];      // 9 KB
  __shared__ __bf16 vls[2][64 * VTP];      // 10 KB
  __shared__ __bf16 plds[8][16 * 32];      // 8 KB (P staging per wave)

  const int tid  = threadIdx.x;
  const int lane = tid & 31;
  const int wave = tid >> 5;

  const int tileid = blockIdx.x * 8 + wave;
  const int b  = tileid >> 7;               // / (SEQ/16)
  const int t  = tileid & 127;
  const int q0 = t * 16;

  const __bf16* qb  = q  + ((size_t)(b * SEQ + q0)) * HD;
  const __bf16* kb  = k  + (size_t)b * SEQ * HD;
  const __bf16* vtb = vt + (size_t)b * HD * SEQ;

  const float SC = 0.125f * 1.44269504088896340736f;  // D^-1/2 * log2(e)
  const int myn = lane & 15;
  const int hi  = (lane & 16) ? 8 : 0;

  // Q fragments live in registers for the whole key loop (D=64 -> 2 frags).
  v16bf qa0 = a_frag_bf16(qb, HD, lane);
  v16bf qa1 = a_frag_bf16(qb + 32, HD, lane);

  v8f acc[4];
  float mrow[8], lrow[8];
  #pragma unroll
  for (int r = 0; r < 8; ++r) { mrow[r] = -1e30f; lrow[r] = 0.0f; }
  #pragma unroll
  for (int i = 0; i < 4; ++i)
    #pragma unroll
    for (int r = 0; r < 8; ++r) acc[i][r] = 0.0f;

  __bf16* pst = plds[wave];
  const int lastkey = q0 + 15;                         // this wave's causal limit
  const int q0_blk  = ((blockIdx.x * 8 + 7) & 127) * 16;  // last wave's q0
  const int nch     = (q0_blk + 16 + 31) >> 5;         // chunks for whole block

  issue_chunk(kb, vtb, 0, kls[0], vls[0], tid);        // prologue fill

  for (int c = 0; c < nch; ++c) {
    const int j  = c * 32;
    const int jn = (c + 1 < nch) ? (j + 32) : j;       // clamp on final iter
    issue_chunk(kb, vtb, jn, kls[(c + 1) & 1], vls[(c + 1) & 1], tid);
    // in-order async completion: <=2 outstanding => chunk c has landed
    asm volatile("s_wait_asynccnt 0x2" ::: "memory");
    __syncthreads();

    if (j <= lastkey) {                                // wave-uniform branch
      const __bf16* kbuf = kls[c & 1];
      const __bf16* vbuf = vls[c & 1];

      // ---- scores: two 16x16 key tiles, K(dim)=64 -> 2 WMMA each ----
      v8f sc0, sc1;
      #pragma unroll
      for (int r = 0; r < 8; ++r) { sc0[r] = 0.0f; sc1[r] = 0.0f; }
      {
        v16bf b0 = b_frag_bf16(kbuf + 0 * KTP,       KTP, lane);
        v16bf b1 = b_frag_bf16(kbuf + 0 * KTP + 32,  KTP, lane);
        sc0 = wmma_bf16(qa0, b0, sc0);
        sc0 = wmma_bf16(qa1, b1, sc0);
        v16bf b2 = b_frag_bf16(kbuf + 16 * KTP,      KTP, lane);
        v16bf b3 = b_frag_bf16(kbuf + 16 * KTP + 32, KTP, lane);
        sc1 = wmma_bf16(qa0, b2, sc1);
        sc1 = wmma_bf16(qa1, b3, sc1);
      }

      // ---- online softmax (exp2 domain), per-row across 16 lanes ----
      #pragma unroll
      for (int r = 0; r < 8; ++r) {
        const int qrow = q0 + r + hi;
        float s0 = sc0[r] * SC;
        float s1 = sc1[r] * SC;
        if (j + myn      > qrow) s0 = -1e30f;      // causal mask
        if (j + 16 + myn > qrow) s1 = -1e30f;
        float mx = fmaxf(s0, s1);
        mx = fmaxf(mx, __shfl_xor(mx, 1, 16));
        mx = fmaxf(mx, __shfl_xor(mx, 2, 16));
        mx = fmaxf(mx, __shfl_xor(mx, 4, 16));
        mx = fmaxf(mx, __shfl_xor(mx, 8, 16));
        const float mn = fmaxf(mrow[r], mx);
        const float p0 = fast_exp2(s0 - mn);
        const float p1 = fast_exp2(s1 - mn);
        float sm = p0 + p1;
        sm += __shfl_xor(sm, 1, 16);
        sm += __shfl_xor(sm, 2, 16);
        sm += __shfl_xor(sm, 4, 16);
        sm += __shfl_xor(sm, 8, 16);
        const float alpha = fast_exp2(mrow[r] - mn);
        lrow[r] = lrow[r] * alpha + sm;
        mrow[r] = mn;
        acc[0][r] *= alpha; acc[1][r] *= alpha;
        acc[2][r] *= alpha; acc[3][r] *= alpha;
        // stage P (C layout -> row-major 16x32 LDS tile)
        const int row = r + hi;
        pst[row * 32 + myn]      = (__bf16)p0;
        pst[row * 32 + 16 + myn] = (__bf16)p1;
      }
      asm volatile("s_wait_dscnt 0" ::: "memory");  // stores visible to A reads

      // ---- P @ V: A = P (16x32 keys), B = vT chunk in LDS, 4 dim tiles ----
      v16bf pa = a_frag_bf16(pst, 32, lane);
      #pragma unroll
      for (int nt = 0; nt < 4; ++nt) {
        v16bf vf = b_frag_bf16(vbuf + (nt * 16) * VTP, VTP, lane);
        acc[nt] = wmma_bf16(pa, vf, acc[nt]);
      }
    }
    __syncthreads();   // all waves done reading buf[c&1] before next overwrite
  }

  // ---- epilogue: out = acc / l, fp32 ----
  float* op = out + ((size_t)(b * SEQ + q0)) * HD;
  #pragma unroll
  for (int r = 0; r < 8; ++r) {
    const float inv = 1.0f / lrow[r];
    #pragma unroll
    for (int nt = 0; nt < 4; ++nt)
      op[(size_t)(r + hi) * HD + nt * 16 + myn] = acc[nt][r] * inv;
  }
}

// ---------------------------------------------------------------------------
extern "C" void kernel_launch(void* const* d_in, const int* in_sizes, int n_in,
                              void* d_out, int out_size, void* d_ws, size_t ws_size,
                              hipStream_t stream) {
  const float* x  = (const float*)d_in[0];
  const float* Wq = (const float*)d_in[1];
  const float* Wk = (const float*)d_in[2];
  const float* Wv = (const float*)d_in[3];
  float* out = (float*)d_out;

  const size_t qkv_elems = (size_t)MROWS * HD;           // 1 Mi elements each
  if (ws_size < 3 * qkv_elems * sizeof(__bf16)) return;  // need 6 MB scratch

  __bf16* qws  = (__bf16*)d_ws;
  __bf16* kws  = qws + qkv_elems;
  __bf16* vtws = kws + qkv_elems;

  // Phase 1: 1024 M-tiles / 8 waves = 128 blocks; z selects Q/K/V.
  qkv_proj_kernel<<<dim3(128, 1, 3), 256, 0, stream>>>(x, Wq, Wk, Wv,
                                                       qws, kws, vtws);
  // Phase 2: 1024 query tiles / 8 waves = 128 blocks.
  attn_kernel<<<dim3(128), 256, 0, stream>>>(qws, kws, vtws, out);
}
